// OPTAttention_56109452755195
// MI455X (gfx1250) — matrix operand order repaired and checked by reading, not compiled, not run on hardware
//
#include <hip/hip_runtime.h>
#include <hip/hip_bf16.h>
#include <stdint.h>

#define Bb  2
#define Tt  2048
#define Dd  2048
#define Hh  32
#define HDd 64

typedef __attribute__((ext_vector_type(16))) _Float16 v16h;
typedef __attribute__((ext_vector_type(8)))  _Float16 h8;
typedef __attribute__((ext_vector_type(4)))  _Float16 h4;
typedef __attribute__((ext_vector_type(8)))  float    v8f;
typedef __attribute__((ext_vector_type(4)))  float    vf4;

__device__ __forceinline__ v8f wmma_f16(v16h a, v16h b, v8f c) {
    return __builtin_amdgcn_wmma_f32_16x16x32_f16(false, a, false, b, (short)0, c, false, false);
}

__device__ __forceinline__ v16h pack16(h8 lo, h8 hi) {
    v16h r;
#pragma unroll
    for (int i = 0; i < 8; i++) { r[i] = lo[i]; r[i + 8] = hi[i]; }
    return r;
}

// A fragment (16x32 f16). Lane pattern: K = (lane/16)*8 + {0..7} and +{16..23}.
__device__ __forceinline__ v16h frag_a_f16(const _Float16* __restrict__ p, int lane) {
    const _Float16* q = p + ((lane >> 4) << 3);
    h8 lo = *(const h8*)(q);
    h8 hi = *(const h8*)(q + 16);
    return pack16(lo, hi);
}

// B fragment (32x16 f16). Lane pattern: K = (lane/16)*8 + {0..7} and +{16..23}.
__device__ __forceinline__ v16h frag_b_f16(const _Float16* __restrict__ p, int lane) {
    const _Float16* q = p + ((lane >> 4) << 3);
    h8 lo = *(const h8*)(q);
    h8 hi = *(const h8*)(q + 16);
    return pack16(lo, hi);
}

__device__ __forceinline__ float rowmax16(float v) {
#pragma unroll
    for (int m = 1; m < 16; m <<= 1) v = fmaxf(v, __shfl_xor(v, m, 32));
    return v;
}
__device__ __forceinline__ float rowsum16(float v) {
#pragma unroll
    for (int m = 1; m < 16; m <<= 1) v += __shfl_xor(v, m, 32);
    return v;
}

// Load 4 elements and return as f16x4 (converts when the source is f32).
__device__ __forceinline__ h4 ld4h(const float* p) {
    vf4 x = *(const vf4*)p;
    h4 r;
#pragma unroll
    for (int i = 0; i < 4; i++) r[i] = (_Float16)x[i];
    return r;
}
__device__ __forceinline__ h4 ld4h(const _Float16* p) { return *(const h4*)p; }

// CDNA5 async global->LDS copy (ASYNCcnt-tracked). vdst = 32-bit LDS byte
// address (low 32 bits of the generic shared-aperture address), vaddr = 64-bit
// global address.
__device__ __forceinline__ void async_ld_b128(unsigned lds_off, const void* g) {
    asm volatile("global_load_async_to_lds_b128 %0, %1, off"
                 :: "v"(lds_off), "v"(g) : "memory");
}
__device__ __forceinline__ void wait_async0() {
    asm volatile("s_wait_asynccnt 0x0" ::: "memory");
}

// ---------------------------------------------------------------------------
// Shared GEMM core: C[128x128] tile per workgroup (8 waves, 32x64 per wave),
// double-buffered f16 LDS staging, k-step 64 (16 WMMAs per wave per step).
// A is [M][K] (f32 or f16), W is nn.Linear weight [N][K] row-major, so
// B[k][n] = W[n][k] and W rows are contiguous along K.
// ---------------------------------------------------------------------------
template <typename TA>
__device__ __forceinline__ void gemm128x128(
    const TA* __restrict__ A, const float* __restrict__ W,
    int m0, int n0,
    _Float16 (*As)[128][64], _Float16 (*Bs)[128][64],
    v8f acc[2][4], int lane, int wv)
{
    const int mband = (wv & 3) * 32;
    const int nband = (wv >> 2) * 64;

    h4 ra[8], rb[8];

    // Prologue: stage k-tile 0
#pragma unroll
    for (int it = 0; it < 8; it++) {
        const int idx = threadIdx.x + it * 256;     // 0..2047 vf4-chunks
        const int row = idx >> 4, c4 = (idx & 15) << 2;
        ra[it] = ld4h(A + (size_t)(m0 + row) * Dd + c4);
        rb[it] = ld4h(W + (size_t)(n0 + row) * Dd + c4);
    }
#pragma unroll
    for (int it = 0; it < 8; it++) {
        const int idx = threadIdx.x + it * 256;
        const int row = idx >> 4, c4 = (idx & 15) << 2;
        *(h4*)&As[0][row][c4] = ra[it];
        *(h4*)&Bs[0][row][c4] = rb[it];
    }
    __syncthreads();

    int cur = 0;
    for (int k0 = 64; k0 <= Dd; k0 += 64) {
        const bool more = (k0 < Dd);
        if (more) {   // issue next tile's global loads before computing
#pragma unroll
            for (int it = 0; it < 8; it++) {
                const int idx = threadIdx.x + it * 256;
                const int row = idx >> 4, c4 = (idx & 15) << 2;
                ra[it] = ld4h(A + (size_t)(m0 + row) * Dd + k0 + c4);
                rb[it] = ld4h(W + (size_t)(n0 + row) * Dd + k0 + c4);
            }
        }

        // Compute on current LDS tiles: 2 k-steps of 32, 8 WMMAs each
#pragma unroll
        for (int kk = 0; kk < 2; kk++) {
            v16h a0 = frag_a_f16(&As[cur][mband + (lane & 15)][kk * 32], lane);
            v16h a1 = frag_a_f16(&As[cur][mband + 16 + (lane & 15)][kk * 32], lane);
            v16h b0 = frag_b_f16(&Bs[cur][nband + (lane & 15)][kk * 32], lane);
            v16h b1 = frag_b_f16(&Bs[cur][nband + 16 + (lane & 15)][kk * 32], lane);
            v16h b2 = frag_b_f16(&Bs[cur][nband + 32 + (lane & 15)][kk * 32], lane);
            v16h b3 = frag_b_f16(&Bs[cur][nband + 48 + (lane & 15)][kk * 32], lane);
            acc[0][0] = wmma_f16(a0, b0, acc[0][0]);
            acc[0][1] = wmma_f16(a0, b1, acc[0][1]);
            acc[0][2] = wmma_f16(a0, b2, acc[0][2]);
            acc[0][3] = wmma_f16(a0, b3, acc[0][3]);
            acc[1][0] = wmma_f16(a1, b0, acc[1][0]);
            acc[1][1] = wmma_f16(a1, b1, acc[1][1]);
            acc[1][2] = wmma_f16(a1, b2, acc[1][2]);
            acc[1][3] = wmma_f16(a1, b3, acc[1][3]);
        }

        if (more) {   // write next tile into the other buffer
#pragma unroll
            for (int it = 0; it < 8; it++) {
                const int idx = threadIdx.x + it * 256;
                const int row = idx >> 4, c4 = (idx & 15) << 2;
                *(h4*)&As[cur ^ 1][row][c4] = ra[it];
                *(h4*)&Bs[cur ^ 1][row][c4] = rb[it];
            }
        }
        __syncthreads();
        cur ^= 1;
    }
}

// ---------------------------------------------------------------------------
// Kernel 1: fused QKV projection -> [B,H,T,HD] f16
// ---------------------------------------------------------------------------
__global__ __launch_bounds__(256) void qkv_gemm_kernel(
    const float* __restrict__ X,
    const float* __restrict__ Wq, const float* __restrict__ bq,
    const float* __restrict__ Wk, const float* __restrict__ bk,
    const float* __restrict__ Wv, const float* __restrict__ bv,
    _Float16* __restrict__ Qb, _Float16* __restrict__ Kb, _Float16* __restrict__ Vb)
{
    __shared__ __align__(16) _Float16 As[2][128][64];
    __shared__ __align__(16) _Float16 Bs[2][128][64];

    const int lane = threadIdx.x & 31;
    const int wv   = threadIdx.x >> 5;

    const float* W; const float* bias; _Float16* Out;
    if (blockIdx.y == 0)      { W = Wq; bias = bq; Out = Qb; }
    else if (blockIdx.y == 1) { W = Wk; bias = bk; Out = Kb; }
    else                      { W = Wv; bias = bv; Out = Vb; }

    const int m0 = (blockIdx.x & 31) * 128;   // 32 M-tiles
    const int n0 = (blockIdx.x >> 5) * 128;   // 16 N-tiles

    v8f acc[2][4];
#pragma unroll
    for (int i = 0; i < 2; i++)
#pragma unroll
        for (int j = 0; j < 4; j++)
#pragma unroll
            for (int r = 0; r < 8; r++) acc[i][j][r] = 0.0f;

    gemm128x128<float>(X, W, m0, n0, As, Bs, acc, lane, wv);

    const int m0w = m0 + (wv & 3) * 32;
    const int n0w = n0 + (wv >> 2) * 64;
#pragma unroll
    for (int j = 0; j < 4; j++) {
        const int   n      = n0w + j * 16 + (lane & 15);
        const float bias_n = bias[n];
        const int   h      = n >> 6;
        const int   hd     = n & 63;
#pragma unroll
        for (int i = 0; i < 2; i++) {
#pragma unroll
            for (int r = 0; r < 8; r++) {
                const int m  = m0w + i * 16 + r + ((lane >> 4) << 3);
                const int bi = m >> 11;
                const int t  = m & (Tt - 1);
                Out[((size_t)((bi * Hh + h) * Tt + t)) * HDd + hd] =
                    (_Float16)(acc[i][j][r] + bias_n);
            }
        }
    }
}

// ---------------------------------------------------------------------------
// Kernel 2: causal flash attention (online softmax), 64 queries per block.
// ---------------------------------------------------------------------------
__global__ __launch_bounds__(128) void flash_attn_kernel(
    const _Float16* __restrict__ Qb, const _Float16* __restrict__ Kb,
    const _Float16* __restrict__ Vb, _Float16* __restrict__ Ctx)
{
    __shared__ __align__(16) _Float16 Qs[64][64];
    __shared__ __align__(16) _Float16 Ks[64][64];
    __shared__ __align__(16) _Float16 Vt[64][64];
    __shared__ __align__(16) _Float16 Ps[4][16][64];

    const int lane = threadIdx.x & 31;
    const int wv   = threadIdx.x >> 5;
    const int qb   = blockIdx.x;
    const int h    = blockIdx.y;
    const int b    = blockIdx.z;

    const size_t hoff = ((size_t)(b * Hh + h)) * Tt * HDd;
    const _Float16* Qh = Qb + hoff;
    const _Float16* Kh = Kb + hoff;
    const _Float16* Vh = Vb + hoff;
    const int qbase = qb * 64;

    for (int i = threadIdx.x; i < 64 * 64; i += 128) {
        const int r = i >> 6, c = i & 63;
        Qs[r][c] = (_Float16)((float)Qh[(size_t)(qbase + r) * HDd + c] * 0.125f);
    }

    const float NEG_INF = -__builtin_inff();
    v8f acc[4];
#pragma unroll
    for (int tv = 0; tv < 4; tv++)
#pragma unroll
        for (int r = 0; r < 8; r++) acc[tv][r] = 0.0f;
    float mrow[8], lrow[8];
#pragma unroll
    for (int r = 0; r < 8; r++) { mrow[r] = NEG_INF; lrow[r] = 0.0f; }

    const unsigned ks_base = (unsigned)(size_t)&Ks[0][0];

    for (int j = 0; j <= qb; j++) {
        __syncthreads();   // previous tiles fully consumed before overwrite

        // K tile: straight f16 copy -> async global->LDS (ASYNCcnt path)
        const _Float16* ksrc = Kh + (size_t)j * 64 * HDd;   // 4096 halfs, contiguous
#pragma unroll
        for (int it = 0; it < 4; it++) {
            const int idx = threadIdx.x + it * 128;          // 512 x 16B chunks
            async_ld_b128(ks_base + idx * 16, ksrc + idx * 8);
        }
        // V tile: transpose to [hd][key] (2-byte granularity, manual)
#pragma unroll
        for (int it = 0; it < 4; it++) {
            const int idx = threadIdx.x + it * 128;
            const int key = idx >> 3, c8 = (idx & 7) * 8;
            h8 v = *(const h8*)(Vh + (size_t)(j * 64 + key) * HDd + c8);
#pragma unroll
            for (int e = 0; e < 8; e++) Vt[c8 + e][key] = v[e];
        }
        wait_async0();
        __syncthreads();

        // S = Q K^T
        v8f s[4];
#pragma unroll
        for (int tn = 0; tn < 4; tn++)
#pragma unroll
            for (int r = 0; r < 8; r++) s[tn][r] = 0.0f;
#pragma unroll
        for (int kk = 0; kk < 2; kk++) {
            v16h a = frag_a_f16(&Qs[wv * 16 + (lane & 15)][kk * 32], lane);
#pragma unroll
            for (int tn = 0; tn < 4; tn++) {
                v16h bm = frag_b_f16(&Ks[tn * 16 + (lane & 15)][kk * 32], lane);
                s[tn] = wmma_f16(a, bm, s[tn]);
            }
        }

        if (j == qb) {   // causal mask on diagonal block
#pragma unroll
            for (int tn = 0; tn < 4; tn++) {
                const int kg = j * 64 + tn * 16 + (lane & 15);
#pragma unroll
                for (int r = 0; r < 8; r++) {
                    const int qg = qbase + wv * 16 + r + ((lane >> 4) << 3);
                    if (kg > qg) s[tn][r] = NEG_INF;
                }
            }
        }

        // Online softmax (rows live in 16-lane halves of the C layout)
#pragma unroll
        for (int r = 0; r < 8; r++) {
            float v = fmaxf(fmaxf(s[0][r], s[1][r]), fmaxf(s[2][r], s[3][r]));
            v = rowmax16(v);
            const float mn   = fmaxf(mrow[r], v);
            const float corr = __expf(mrow[r] - mn);
            mrow[r] = mn;
            float sum = 0.0f;
#pragma unroll
            for (int tn = 0; tn < 4; tn++) {
                const float p = __expf(s[tn][r] - mn);
                s[tn][r] = p;
                sum += p;
            }
            sum = rowsum16(sum);
            lrow[r] = lrow[r] * corr + sum;
#pragma unroll
            for (int tv = 0; tv < 4; tv++) acc[tv][r] *= corr;
        }

        // P: C layout -> LDS -> A layout (wave-private; LDS in-order per wave)
#pragma unroll
        for (int tn = 0; tn < 4; tn++)
#pragma unroll
            for (int r = 0; r < 8; r++) {
                const int row = r + ((lane >> 4) << 3);
                Ps[wv][row][tn * 16 + (lane & 15)] = (_Float16)s[tn][r];
            }

        // O += P V
#pragma unroll
        for (int kk = 0; kk < 2; kk++) {
            v16h a = frag_a_f16(&Ps[wv][lane & 15][kk * 32], lane);
#pragma unroll
            for (int tv = 0; tv < 4; tv++) {
                v16h bm = frag_b_f16(&Vt[tv * 16 + (lane & 15)][kk * 32], lane);
                acc[tv] = wmma_f16(a, bm, acc[tv]);
            }
        }
    }

    // Finalize: O /= l, write ctx as [b][t][h*64+hd] f16
#pragma unroll
    for (int r = 0; r < 8; r++) {
        const float inv = 1.0f / lrow[r];
        const int row = r + ((lane >> 4) << 3);
        const int t   = qbase + wv * 16 + row;
#pragma unroll
        for (int tv = 0; tv < 4; tv++) {
            const int hd = tv * 16 + (lane & 15);
            Ctx[((size_t)(b * Tt + t)) * Dd + h * HDd + hd] =
                (_Float16)(acc[tv][r] * inv);
        }
    }
}

// ---------------------------------------------------------------------------
// Kernel 3: output projection. out = ctx @ Wo^T + bo (f32 dense output)
// ---------------------------------------------------------------------------
__global__ __launch_bounds__(256) void out_gemm_kernel(
    const _Float16* __restrict__ Xh, const float* __restrict__ W,
    const float* __restrict__ bias, float* __restrict__ Out)
{
    __shared__ __align__(16) _Float16 As[2][128][64];
    __shared__ __align__(16) _Float16 Bs[2][128][64];

    const int lane = threadIdx.x & 31;
    const int wv   = threadIdx.x >> 5;
    const int m0 = (blockIdx.x & 31) * 128;
    const int n0 = (blockIdx.x >> 5) * 128;

    v8f acc[2][4];
#pragma unroll
    for (int i = 0; i < 2; i++)
#pragma unroll
        for (int j = 0; j < 4; j++)
#pragma unroll
            for (int r = 0; r < 8; r++) acc[i][j][r] = 0.0f;

    gemm128x128<_Float16>(Xh, W, m0, n0, As, Bs, acc, lane, wv);

    const int m0w = m0 + (wv & 3) * 32;
    const int n0w = n0 + (wv >> 2) * 64;
#pragma unroll
    for (int j = 0; j < 4; j++) {
        const int   n      = n0w + j * 16 + (lane & 15);
        const float bias_n = bias[n];
#pragma unroll
        for (int i = 0; i < 2; i++) {
#pragma unroll
            for (int r = 0; r < 8; r++) {
                const int m = m0w + i * 16 + r + ((lane >> 4) << 3);
                Out[(size_t)m * Dd + n] = acc[i][j][r] + bias_n;
            }
        }
    }
}

// ---------------------------------------------------------------------------
extern "C" void kernel_launch(void* const* d_in, const int* in_sizes, int n_in,
                              void* d_out, int out_size, void* d_ws, size_t ws_size,
                              hipStream_t stream) {
    const float* X  = (const float*)d_in[0];
    const float* Wq = (const float*)d_in[1];
    const float* bq = (const float*)d_in[2];
    const float* Wk = (const float*)d_in[3];
    const float* bk = (const float*)d_in[4];
    const float* Wv = (const float*)d_in[5];
    const float* bv = (const float*)d_in[6];
    const float* Wo = (const float*)d_in[7];
    const float* bo = (const float*)d_in[8];

    const size_t QE = (size_t)Bb * Hh * Tt * HDd;   // 8388608 elements
    _Float16* Qb  = (_Float16*)d_ws;
    _Float16* Kb  = Qb + QE;
    _Float16* Vb  = Kb + QE;
    _Float16* Ctx = Vb + QE;

    qkv_gemm_kernel<<<dim3(512, 3), 256, 0, stream>>>(
        X, Wq, bq, Wk, bk, Wv, bv, Qb, Kb, Vb);

    flash_attn_kernel<<<dim3(Tt / 64, Hh, Bb), 128, 0, stream>>>(Qb, Kb, Vb, Ctx);

    out_gemm_kernel<<<dim3(512), 256, 0, stream>>>(Ctx, Wo, bo, (float*)d_out);
}
